// PhpNetGraphDependenceTokens_62010737820206
// MI455X (gfx1250) — matrix operand order repaired and checked by reading, not compiled, not run on hardware
//
#include <hip/hip_runtime.h>
#include <stdint.h>

// ---------------- problem constants ----------------
#define FEAT    2000
#define FPAD    2048      // padded feature/K dim (mult of 32)
#define NNODES  2048
#define NEDGES  8192
#define NGRAPHS 16
#define TOKENS  20
#define EMBDIM  100

typedef __attribute__((ext_vector_type(16))) __bf16 v16bf;
typedef __attribute__((ext_vector_type(8)))  float  v8f;
typedef unsigned long long u64;
typedef unsigned int       u32;
typedef unsigned short     u16;

// ---------------- helpers ----------------
__device__ __forceinline__ u32 f2ord(float f) {            // monotone float->uint
    u32 b = __float_as_uint(f);
    return (b & 0x80000000u) ? ~b : (b | 0x80000000u);
}
__device__ __forceinline__ float ord2f(u32 o) {
    u32 b = (o & 0x80000000u) ? (o & 0x7fffffffu) : ~o;
    return __uint_as_float(b);
}
__device__ __forceinline__ u16 f2bf(float f) {             // fp32 -> bf16 RNE
    u32 b = __float_as_uint(f);
    u32 r = b + 0x7fffu + ((b >> 16) & 1u);
    return (u16)(r >> 16);
}
// ord(-FLT_MAX) on host: bits(-FLT_MAX)=0xFF7FFFFF, negative -> ~bits = 0x00800000
#define ORD_NEG_FLT_MAX 0x00800000u

// ---------------- generic fills ----------------
__global__ void k_fill_f32(float* p, float v, int n) {
    int i = blockIdx.x * blockDim.x + threadIdx.x; if (i < n) p[i] = v;
}
__global__ void k_fill_u32(u32* p, u32 v, int n) {
    int i = blockIdx.x * blockDim.x + threadIdx.x; if (i < n) p[i] = v;
}

// ---------------- embedding gather: X[n, t*100+e] = emb[tok[n,t], e], pad cols zero --------
__global__ void k_embed(const int* __restrict__ tok, const float* __restrict__ emb,
                        float* __restrict__ X) {
    int i = blockIdx.x * blockDim.x + threadIdx.x;       // over NNODES*FPAD
    int n = i >> 11, c = i & 2047;
    float v = 0.f;
    if (c < FEAT) {
        int tk = tok[n * TOKENS + c / EMBDIM];
        v = emb[(size_t)tk * EMBDIM + c % EMBDIM];
    }
    X[i] = v;
}

// ---------------- fp32 (already zero-padded) -> bf16 ----------------
__global__ void k_tobf16(const float* __restrict__ src, u16* __restrict__ dst, int n) {
    int i = blockIdx.x * blockDim.x + threadIdx.x; if (i < n) dst[i] = f2bf(src[i]);
}

// ---------------- pack weight W (Ksrc x Nsrc row-major) into B-tile layout -------------
// Bp[((n*kt + k/32)*32 + k%32)] = bf16(W[k,n]); linear dst index == n*Kpad + k.
__global__ void k_pack_B(const float* __restrict__ W, u16* __restrict__ Bp,
                         int Ksrc, int Nsrc, int Kpad, int Npad) {
    int i = blockIdx.x * blockDim.x + threadIdx.x;       // over Npad*Kpad
    int n = i / Kpad, k = i % Kpad;
    float v = (k < Ksrc && n < Nsrc) ? W[(size_t)k * Nsrc + n] : 0.f;
    Bp[i] = f2bf(v);
}

// ---------------- WMMA GEMM: C(f32, MxN) = A(bf16, MxK row-major) @ Bpacked ------------
// register-blocked: one wave computes an (MT*16) x (NT*16) block; K multiple of 32.
// grid = (Npad/(NT*16), Mpad/(MT*16)), block = 32.  Stores guarded by col < Nreal.
// __launch_bounds__(32, 1): single wave per block, allow full VGPR budget (no acc spills).
template <int MT, int NT>
__global__ __launch_bounds__(32, 1)
void k_wmma_gemm_t(const u16* __restrict__ A, const u16* __restrict__ Bp,
                   float* __restrict__ C, int K, int lda, int ldc, int Nreal) {
    const int lane = threadIdx.x;
    const int m16  = lane & 15;
    const int h    = lane >> 4;
    const int bN   = blockIdx.x * (NT * 16);
    const int bM   = blockIdx.y * (MT * 16);
    const int kt   = K >> 5;

    union U { u32 u[8]; v16bf v; };

    v8f acc[MT][NT];
#pragma unroll
    for (int i = 0; i < MT; ++i)
#pragma unroll
        for (int j = 0; j < NT; ++j) acc[i][j] = {};

    for (int kb = 0; kb < kt; ++kb) {
        const int k0 = kb << 5;
        U ua[MT], ub[NT];
#pragma unroll
        for (int i = 0; i < MT; ++i) {
            const u16* ab = A + (size_t)(bM + i * 16 + m16) * lda + k0 + h * 8;
#pragma unroll
            for (int g = 0; g < 4; ++g) ua[i].u[g]     = *(const u32*)(ab + 2 * g);
#pragma unroll
            for (int g = 0; g < 4; ++g) ua[i].u[4 + g] = *(const u32*)(ab + 16 + 2 * g);
        }
#pragma unroll
        for (int j = 0; j < NT; ++j) {
            const u16* bb = Bp + ((size_t)(bN + j * 16 + m16) * kt + kb) * 32 + h * 8;
#pragma unroll
            for (int g = 0; g < 4; ++g) ub[j].u[g]     = *(const u32*)(bb + 2 * g);
#pragma unroll
            for (int g = 0; g < 4; ++g) ub[j].u[4 + g] = *(const u32*)(bb + 16 + 2 * g);
        }
#pragma unroll
        for (int i = 0; i < MT; ++i)
#pragma unroll
            for (int j = 0; j < NT; ++j)
                acc[i][j] = __builtin_amdgcn_wmma_f32_16x16x32_bf16(
                    false, ua[i].v, false, ub[j].v, (short)0, acc[i][j], false, false);
    }
#pragma unroll
    for (int i = 0; i < MT; ++i)
#pragma unroll
        for (int j = 0; j < NT; ++j) {
            int col = bN + j * 16 + m16;
            if (col < Nreal) {
                float* crow = C + (size_t)(bM + i * 16 + 8 * h) * ldc + col;
#pragma unroll
                for (int r = 0; r < 8; ++r) crow[(size_t)r * ldc] = acc[i][j][r];
            }
        }
}

// ---------------- GCN norm + aggregation ----------------
__global__ void k_deg(const int* __restrict__ S, const int* __restrict__ T,
                      const int* __restrict__ EV, float* __restrict__ deg, int E) {
    int e = blockIdx.x * blockDim.x + threadIdx.x;
    if (e < E) { int s = S[e], t = T[e]; if (EV[e] && s != t) atomicAdd(&deg[s], 1.0f); }
}
__global__ void k_rsqrt_ip(float* __restrict__ d, int n) {
    int i = blockIdx.x * blockDim.x + threadIdx.x; if (i < n) d[i] = rsqrtf(d[i]);
}
// self-loop term + bias: Y = XW * dinv^2 + b   (pad cols -> 0)
__global__ void k_self(const float* __restrict__ XW, const float* __restrict__ dinv,
                       const float* __restrict__ b, float* __restrict__ Y) {
    int i = blockIdx.x * blockDim.x + threadIdx.x;       // NNODES*FPAD
    int n = i >> 11, f = i & 2047;
    float di = dinv[n];
    Y[i] = (f < FEAT) ? (XW[i] * di * di + b[f]) : 0.f;
}
__global__ void k_edge_agg(const int* __restrict__ S, const int* __restrict__ T,
                           const int* __restrict__ EV, const float* __restrict__ XW,
                           const float* __restrict__ dinv, float* __restrict__ Y) {
    int e = blockIdx.y;
    int f = blockIdx.x * blockDim.x + threadIdx.x;
    if (f >= FEAT) return;
    int s = S[e], t = T[e];
    if (!EV[e] || s == t) return;
    float c = dinv[s] * dinv[t];
    atomicAdd(&Y[(size_t)t * FPAD + f], XW[(size_t)s * FPAD + f] * c);
}

// ---------------- edge pooling ----------------
// raw[e] = dot(x[s],Wp[0:2000]) + dot(x[t],Wp[2000:4000]) + bp
__global__ void k_pool_raw(const int* __restrict__ S, const int* __restrict__ T,
                           const float* __restrict__ X, const float* __restrict__ Wp,
                           const float* __restrict__ bp, float* __restrict__ raw) {
    __shared__ float red[256];
    int e = blockIdx.x, tid = threadIdx.x;
    int s = S[e], t = T[e];
    const float* xs = X + (size_t)s * FPAD;
    const float* xt = X + (size_t)t * FPAD;
    float acc = 0.f;
    for (int f = tid; f < FEAT; f += 256)
        acc += xs[f] * Wp[f] + xt[f] * Wp[FEAT + f];
    red[tid] = acc; __syncthreads();
    for (int o = 128; o > 0; o >>= 1) { if (tid < o) red[tid] += red[tid + o]; __syncthreads(); }
    if (tid == 0) raw[e] = red[0] + bp[0];
}
__global__ void k_segmax(const int* __restrict__ T, const int* __restrict__ EV,
                         const float* __restrict__ raw, u32* __restrict__ mx, int E) {
    int e = blockIdx.x * blockDim.x + threadIdx.x;
    if (e < E && EV[e]) atomicMax(&mx[T[e]], f2ord(raw[e]));
}
__global__ void k_mxfix(u32* __restrict__ mx, int n) {   // in-place u32 -> f32
    int i = blockIdx.x * blockDim.x + threadIdx.x;
    if (i < n) { u32 o = mx[i]; float v = o ? ord2f(o) : 0.f; ((float*)mx)[i] = v; }
}
__global__ void k_ex(const int* __restrict__ T, const int* __restrict__ EV,
                     const float* __restrict__ raw, const float* __restrict__ mx,
                     float* __restrict__ ex, float* __restrict__ den, int E) {
    int e = blockIdx.x * blockDim.x + threadIdx.x;
    if (e >= E) return;
    float v = 0.f;
    if (EV[e]) { v = expf(raw[e] - mx[T[e]]); atomicAdd(&den[T[e]], v); }
    ex[e] = v;
}
__global__ void k_score_keys(const int* __restrict__ T, const int* __restrict__ EV,
                             const float* __restrict__ ex, const float* __restrict__ den,
                             float* __restrict__ score, u64* __restrict__ keys, int E) {
    int e = blockIdx.x * blockDim.x + threadIdx.x;
    if (e >= E) return;
    float sc = 0.f;
    if (EV[e]) { float d = den[T[e]]; sc = ex[e] / fmaxf(d, 1e-30f) + 0.5f; }
    score[e] = sc;
    float srt = EV[e] ? sc : -1.0f;                      // sort descending by srt, stable
    keys[e] = ((u64)f2ord(-srt) << 13) | (u32)e;         // ascending sort == desc score
}
// single-block: bitonic sort keys then serial greedy matching (PyG EdgePooling loop)
__global__ void k_sort_greedy(const u64* __restrict__ keys, const int* __restrict__ S,
                              const int* __restrict__ T, const int* __restrict__ EV,
                              const int* __restrict__ NV, int* __restrict__ chosen) {
    extern __shared__ u64 sk[];                          // NEDGES entries
    __shared__ unsigned char merged[NNODES];
    int tid = threadIdx.x;
    for (int i = tid; i < NEDGES; i += blockDim.x) sk[i] = keys[i];
    for (int i = tid; i < NNODES; i += blockDim.x) merged[i] = NV[i] ? 0 : 1;
    __syncthreads();
    for (int kk = 2; kk <= NEDGES; kk <<= 1)
        for (int j = kk >> 1; j > 0; j >>= 1) {
            for (int i = tid; i < NEDGES; i += blockDim.x) {
                int ixj = i ^ j;
                if (ixj > i) {
                    bool up = ((i & kk) == 0);
                    u64 a = sk[i], b = sk[ixj];
                    if ((a > b) == up) { sk[i] = b; sk[ixj] = a; }
                }
            }
            __syncthreads();
        }
    if (tid == 0) {
        for (int j = 0; j < NEDGES; ++j) {
            int e = (int)(sk[j] & 8191u);
            int s = S[e], t = T[e];
            int ok = (EV[e] && !merged[s] && !merged[t]) ? 1 : 0;
            if (ok) { merged[s] = 1; merged[t] = 1; }
            chosen[e] = ok;
        }
    }
}
__global__ void k_cluster_init(int* __restrict__ cl, float* __restrict__ csc, int n) {
    int i = blockIdx.x * blockDim.x + threadIdx.x;
    if (i < n) { cl[i] = i; csc[i] = 1.0f; }
}
__global__ void k_cluster_set(const int* __restrict__ S, const int* __restrict__ T,
                              const int* __restrict__ chosen, const float* __restrict__ score,
                              int* __restrict__ cl, float* __restrict__ csc, int E) {
    int e = blockIdx.x * blockDim.x + threadIdx.x;
    if (e >= E || !chosen[e]) return;
    int s = S[e], t = T[e], m = s < t ? s : t;
    cl[s] = m; cl[t] = m; csc[m] = score[e];
}
__global__ void k_scatter(const float* __restrict__ Y, const int* __restrict__ cl,
                          float* __restrict__ X) {      // X pre-zeroed
    int n = blockIdx.y;
    int f = blockIdx.x * blockDim.x + threadIdx.x;
    if (f >= FEAT) return;
    atomicAdd(&X[(size_t)cl[n] * FPAD + f], Y[(size_t)n * FPAD + f]);
}
__global__ void k_scale_relu(float* __restrict__ X, const float* __restrict__ csc) {
    int i = blockIdx.x * blockDim.x + threadIdx.x;       // NNODES*FPAD
    int n = i >> 11;
    X[i] = fmaxf(X[i] * csc[n], 0.f);
}
__global__ void k_nv_upd(int* __restrict__ NV, const int* __restrict__ cl, int n) {
    int i = blockIdx.x * blockDim.x + threadIdx.x;
    if (i < n) NV[i] = (NV[i] && cl[i] == i) ? 1 : 0;
}
__global__ void k_remap_keys(int* __restrict__ S, int* __restrict__ T,
                             const int* __restrict__ EV, const int* __restrict__ cl,
                             u64* __restrict__ keys, int E) {
    int e = blockIdx.x * blockDim.x + threadIdx.x;
    if (e >= E) return;
    int ns = cl[S[e]], nt = cl[T[e]];
    S[e] = ns; T[e] = nt;
    u64 kv = EV[e] ? ((u64)ns * (u64)NNODES + (u64)nt) : (u64)NNODES * (u64)NNODES;
    keys[e] = (kv << 13) | (u32)e;
}
__global__ void k_sort_dedup(const u64* __restrict__ keys, int* __restrict__ EV) {
    extern __shared__ u64 sk[];
    int tid = threadIdx.x;
    for (int i = tid; i < NEDGES; i += blockDim.x) sk[i] = keys[i];
    __syncthreads();
    for (int kk = 2; kk <= NEDGES; kk <<= 1)
        for (int j = kk >> 1; j > 0; j >>= 1) {
            for (int i = tid; i < NEDGES; i += blockDim.x) {
                int ixj = i ^ j;
                if (ixj > i) {
                    bool up = ((i & kk) == 0);
                    u64 a = sk[i], b = sk[ixj];
                    if ((a > b) == up) { sk[i] = b; sk[ixj] = a; }
                }
            }
            __syncthreads();
        }
    for (int i = tid; i < NEDGES; i += blockDim.x) {
        u64 cur = sk[i]; u64 kv = cur >> 13; int e = (int)(cur & 8191u);
        int first = (i == 0) || ((sk[i - 1] >> 13) != kv);
        EV[e] = (first && kv < (u64)NNODES * (u64)NNODES) ? 1 : 0;
    }
}

// ---------------- global max pool ----------------
__global__ void k_gmax(const float* __restrict__ X, const int* __restrict__ batch,
                       const int* __restrict__ NV, u32* __restrict__ G) {
    int n = blockIdx.y;
    int f = blockIdx.x * blockDim.x + threadIdx.x;
    if (f >= FEAT || !NV[n]) return;
    atomicMax(&G[(size_t)batch[n] * FEAT + f], f2ord(X[(size_t)n * FPAD + f]));
}
__global__ void k_unord(u32* __restrict__ G, int n) {    // in-place u32 -> f32
    int i = blockIdx.x * blockDim.x + threadIdx.x;
    if (i < n) ((float*)G)[i] = ord2f(G[i]);
}

// ---------------- MLP head ----------------
__global__ void k_build_hbf(const float* __restrict__ G1, const float* __restrict__ G2,
                            u16* __restrict__ H) {       // 16 x 4096 bf16, pad zero
    int i = blockIdx.x * blockDim.x + threadIdx.x;       // 16*4096
    int r = i >> 12, c = i & 4095;
    float v = 0.f;
    if (c < 2000)      v = G1[(size_t)r * FEAT + c];
    else if (c < 4000) v = G2[(size_t)r * FEAT + (c - 2000)];
    H[i] = f2bf(v);
}
__global__ void k_bn_relu_bf(const float* __restrict__ Z, const float* __restrict__ lb,
                             const float* __restrict__ bg, const float* __restrict__ bb,
                             const float* __restrict__ bm, const float* __restrict__ bv,
                             u16* __restrict__ dst, int cols, int ld) {
    int i = blockIdx.x * blockDim.x + threadIdx.x;       // 16*ld
    int c = i % ld;
    float z = 0.f;
    if (c < cols) {
        z = Z[i] + lb[c];
        z = (z - bm[c]) * rsqrtf(bv[c] + 1e-5f) * bg[c] + bb[c];
        z = fmaxf(z, 0.f);
    }
    dst[i] = f2bf(z);
}
__global__ void k_bn_relu_ip(float* __restrict__ Z, const float* __restrict__ lb,
                             const float* __restrict__ bg, const float* __restrict__ bb,
                             const float* __restrict__ bm, const float* __restrict__ bv,
                             int cols) {
    int i = blockIdx.x * blockDim.x + threadIdx.x;       // 16*cols
    int c = i % cols;
    float z = Z[i] + lb[c];
    z = (z - bm[c]) * rsqrtf(bv[c] + 1e-5f) * bg[c] + bb[c];
    Z[i] = fmaxf(z, 0.f);
}
__global__ void k_final(const float* __restrict__ H2, const float* __restrict__ W,
                        const float* __restrict__ b, float* __restrict__ out) {
    int tid = threadIdx.x;                               // 64 threads
    if (tid >= 64) return;
    int r = tid >> 2, c = tid & 3;
    float acc = 0.f;
    for (int k = 0; k < 1600; ++k) acc += H2[(size_t)r * 1600 + k] * W[(size_t)k * 4 + c];
    out[tid] = fmaxf(acc + b[c], 0.f);
}

// =====================================================================================
//                                    host driver
// =====================================================================================
// workspace layout (bytes)
#define O_X     ((size_t)0)                       // 2048*2048*4
#define O_XW    (O_X   + (size_t)16777216)
#define O_Y     (O_XW  + (size_t)16777216)
#define O_XBF   (O_Y   + (size_t)16777216)        // 2048*2048*2
#define O_WBF   (O_XBF + (size_t)8388608)         // 3008*4096*2 (reused for all weights)
#define O_HBF   (O_WBF + (size_t)24641536)        // 16*4096*2
#define O_H1    (O_HBF + (size_t)131072)          // 16*3008*4
#define O_H2    (O_H1  + (size_t)192512)          // 16*1600*4
#define O_G1    (O_H2  + (size_t)102400)          // 16*2000*4
#define O_G2    (O_G1  + (size_t)128000)
#define O_S     (O_G2  + (size_t)128000)          // 8192*4 (+8192*4 for T right after)
#define O_T     (O_S   + (size_t)32768)
#define O_EV    (O_T   + (size_t)32768)
#define O_NV    (O_EV  + (size_t)32768)
#define O_DEG   (O_NV  + (size_t)8192)
#define O_RAW   (O_DEG + (size_t)8192)
#define O_EX    (O_RAW + (size_t)32768)
#define O_SCORE (O_EX  + (size_t)32768)
#define O_MX    (O_SCORE+(size_t)32768)
#define O_DEN   (O_MX  + (size_t)8192)
#define O_KEYS  (O_DEN + (size_t)8192)            // 8192*8
#define O_CHOSE (O_KEYS+ (size_t)65536)
#define O_CLUS  (O_CHOSE+(size_t)32768)
#define O_CSC   (O_CLUS+ (size_t)8192)

static void run_branch(const float* emb, const float* const* convW, const float* const* convb,
                       const float* const* poolW, const float* const* poolb,
                       const int* tok, const int* ei, const int* batch,
                       u32* G /* 16*2000 out, ordered-u32 then f32 */, char* ws,
                       hipStream_t stream) {
    float* X   = (float*)(ws + O_X);
    float* XW  = (float*)(ws + O_XW);
    float* Y   = (float*)(ws + O_Y);
    u16*   XBF = (u16*)  (ws + O_XBF);
    u16*   WBF = (u16*)  (ws + O_WBF);
    int*   S   = (int*)  (ws + O_S);
    int*   T   = (int*)  (ws + O_T);
    int*   EV  = (int*)  (ws + O_EV);
    int*   NV  = (int*)  (ws + O_NV);
    float* DEG = (float*)(ws + O_DEG);
    float* RAW = (float*)(ws + O_RAW);
    float* EX  = (float*)(ws + O_EX);
    float* SC  = (float*)(ws + O_SCORE);
    u32*   MX  = (u32*)  (ws + O_MX);
    float* DEN = (float*)(ws + O_DEN);
    u64*   KEY = (u64*)  (ws + O_KEYS);
    int*   CH  = (int*)  (ws + O_CHOSE);
    int*   CL  = (int*)  (ws + O_CLUS);
    float* CSC = (float*)(ws + O_CSC);

    const int NM = (NNODES * FPAD) / 256;                // 16384 blocks elementwise
    const int EB = (NEDGES + 255) / 256;
    const int NB = (NNODES + 255) / 256;

    (void)hipMemcpyAsync(S, ei, 2 * NEDGES * sizeof(int), hipMemcpyDeviceToDevice, stream);
    k_fill_u32<<<EB, 256, 0, stream>>>((u32*)EV, 1u, NEDGES);
    k_fill_u32<<<NB, 256, 0, stream>>>((u32*)NV, 1u, NNODES);
    k_embed<<<NM, 256, 0, stream>>>(tok, emb, X);

    for (int L = 0; L < 4; ++L) {
        // ---- GCNConv: XW = bf16(X) @ bf16(W) via WMMA (64x64 register-blocked waves)
        k_tobf16<<<NM, 256, 0, stream>>>(X, XBF, NNODES * FPAD);
        k_pack_B<<<(2048 * FPAD) / 256, 256, 0, stream>>>(convW[L], WBF, FEAT, FEAT, FPAD, 2048);
        k_wmma_gemm_t<4, 4><<<dim3(2048 / 64, NNODES / 64), 32, 0, stream>>>(
            XBF, WBF, XW, FPAD, FPAD, FPAD, FEAT);
        // ---- symmetric norm + aggregate
        k_fill_f32<<<NB, 256, 0, stream>>>(DEG, 1.0f, NNODES);       // +1 self loop
        k_deg<<<EB, 256, 0, stream>>>(S, T, EV, DEG, NEDGES);
        k_rsqrt_ip<<<NB, 256, 0, stream>>>(DEG, NNODES);             // DEG := dinv
        k_self<<<NM, 256, 0, stream>>>(XW, DEG, convb[L], Y);
        k_edge_agg<<<dim3(8, NEDGES), 256, 0, stream>>>(S, T, EV, XW, DEG, Y);
        // ---- EdgePooling on Y
        k_pool_raw<<<NEDGES, 256, 0, stream>>>(S, T, Y, poolW[L], poolb[L], RAW);
        (void)hipMemsetAsync(MX, 0, NNODES * sizeof(u32), stream);
        k_segmax<<<EB, 256, 0, stream>>>(T, EV, RAW, MX, NEDGES);
        k_mxfix<<<NB, 256, 0, stream>>>(MX, NNODES);
        (void)hipMemsetAsync(DEN, 0, NNODES * sizeof(float), stream);
        k_ex<<<EB, 256, 0, stream>>>(T, EV, RAW, (float*)MX, EX, DEN, NEDGES);
        k_score_keys<<<EB, 256, 0, stream>>>(T, EV, EX, DEN, SC, KEY, NEDGES);
        k_sort_greedy<<<1, 1024, NEDGES * sizeof(u64), stream>>>(KEY, S, T, EV, NV, CH);
        k_cluster_init<<<NB, 256, 0, stream>>>(CL, CSC, NNODES);
        k_cluster_set<<<EB, 256, 0, stream>>>(S, T, CH, SC, CL, CSC, NEDGES);
        (void)hipMemsetAsync(X, 0, (size_t)NNODES * FPAD * sizeof(float), stream);
        k_scatter<<<dim3(8, NNODES), 256, 0, stream>>>(Y, CL, X);
        k_scale_relu<<<NM, 256, 0, stream>>>(X, CSC);
        k_nv_upd<<<NB, 256, 0, stream>>>(NV, CL, NNODES);
        k_remap_keys<<<EB, 256, 0, stream>>>(S, T, EV, CL, KEY, NEDGES);
        k_sort_dedup<<<1, 1024, NEDGES * sizeof(u64), stream>>>(KEY, EV);
    }
    // ---- global max pool (masked by node validity)
    k_fill_u32<<<(NGRAPHS * FEAT + 255) / 256, 256, 0, stream>>>(G, ORD_NEG_FLT_MAX, NGRAPHS * FEAT);
    k_gmax<<<dim3(8, NNODES), 256, 0, stream>>>(X, batch, NV, G);
    k_unord<<<(NGRAPHS * FEAT + 255) / 256, 256, 0, stream>>>(G, NGRAPHS * FEAT);
}

// input leaf indices (params flattened in JAX pytree order: dict keys sorted, lists expanded)
enum {
    I_BN01_B = 0, I_BN01_G, I_BN01_M, I_BN01_V,
    I_BN02_B, I_BN02_G, I_BN02_M, I_BN02_V,
    I_CONVW1 = 8,  I_CONVW2 = 12, I_CONVB1 = 16, I_CONVB2 = 20,
    I_EMB1 = 24, I_EMB2 = 25,
    I_LIN01_W = 26, I_LIN01_B = 27, I_LIN02_W = 28, I_LIN02_B = 29,
    I_LIN1_W = 30, I_LIN1_B = 31,
    I_POOLW1 = 32, I_POOLW2 = 36, I_POOLB1 = 40, I_POOLB2 = 44,
    I_XPD = 48, I_EIPD = 49, I_BPD = 50, I_XCF = 51, I_EICF = 52, I_BCF = 53
};

extern "C" void kernel_launch(void* const* d_in, const int* in_sizes, int n_in,
                              void* d_out, int out_size, void* d_ws, size_t ws_size,
                              hipStream_t stream) {
    (void)in_sizes; (void)n_in; (void)out_size; (void)ws_size;   // needs ~85 MB workspace
    char* ws = (char*)d_ws;

    const float* convW1[4], *convW2[4], *convb1[4], *convb2[4];
    const float* poolW1[4], *poolW2[4], *poolb1[4], *poolb2[4];
    for (int i = 0; i < 4; ++i) {
        convW1[i] = (const float*)d_in[I_CONVW1 + i];
        convW2[i] = (const float*)d_in[I_CONVW2 + i];
        convb1[i] = (const float*)d_in[I_CONVB1 + i];
        convb2[i] = (const float*)d_in[I_CONVB2 + i];
        poolW1[i] = (const float*)d_in[I_POOLW1 + i];
        poolW2[i] = (const float*)d_in[I_POOLW2 + i];
        poolb1[i] = (const float*)d_in[I_POOLB1 + i];
        poolb2[i] = (const float*)d_in[I_POOLB2 + i];
    }
    u32* G1 = (u32*)(ws + O_G1);
    u32* G2 = (u32*)(ws + O_G2);

    run_branch((const float*)d_in[I_EMB1], convW1, convb1, poolW1, poolb1,
               (const int*)d_in[I_XPD], (const int*)d_in[I_EIPD], (const int*)d_in[I_BPD],
               G1, ws, stream);
    run_branch((const float*)d_in[I_EMB2], convW2, convb2, poolW2, poolb2,
               (const int*)d_in[I_XCF], (const int*)d_in[I_EICF], (const int*)d_in[I_BCF],
               G2, ws, stream);

    // ---- MLP head: h(16x4000) -> 3000 -> 1600 -> 4
    u16*   HBF = (u16*)  (ws + O_HBF);
    u16*   WBF = (u16*)  (ws + O_WBF);
    float* H1  = (float*)(ws + O_H1);
    float* H2  = (float*)(ws + O_H2);

    k_build_hbf<<<(16 * 4096) / 256, 256, 0, stream>>>((const float*)G1, (const float*)G2, HBF);
    k_pack_B<<<(3008 * 4096) / 256, 256, 0, stream>>>((const float*)d_in[I_LIN01_W], WBF,
                                                      4000, 3000, 4096, 3008);
    k_wmma_gemm_t<1, 4><<<dim3(3008 / 64, 1), 32, 0, stream>>>(HBF, WBF, H1, 4096, 4096, 3008, 3000);
    k_bn_relu_bf<<<(16 * 3008) / 256, 256, 0, stream>>>(H1, (const float*)d_in[I_LIN01_B],
        (const float*)d_in[I_BN01_G], (const float*)d_in[I_BN01_B],
        (const float*)d_in[I_BN01_M], (const float*)d_in[I_BN01_V], HBF, 3000, 3008);
    k_pack_B<<<(3008 * 1600) / 256, 256, 0, stream>>>((const float*)d_in[I_LIN02_W], WBF,
                                                      3000, 1600, 3008, 1600);
    k_wmma_gemm_t<1, 4><<<dim3(1600 / 64, 1), 32, 0, stream>>>(HBF, WBF, H2, 3008, 3008, 1600, 1600);
    k_bn_relu_ip<<<(16 * 1600) / 256, 256, 0, stream>>>(H2, (const float*)d_in[I_LIN02_B],
        (const float*)d_in[I_BN02_G], (const float*)d_in[I_BN02_B],
        (const float*)d_in[I_BN02_M], (const float*)d_in[I_BN02_V], 1600);
    k_final<<<1, 64, 0, stream>>>(H2, (const float*)d_in[I_LIN1_W],
                                  (const float*)d_in[I_LIN1_B], (float*)d_out);
}
